// CrossAttentionBlock_78314433675872
// MI455X (gfx1250) — matrix operand order repaired and checked by reading, not compiled, not run on hardware
//
#include <hip/hip_runtime.h>
#include <hip/hip_bf16.h>

// CDNA5 / gfx1250. Wave32. WMMA bf16 16x16x32.
typedef __attribute__((ext_vector_type(16))) __bf16 v16bf;
typedef __attribute__((ext_vector_type(8)))  float  v8f;
typedef __attribute__((ext_vector_type(2)))  float  f32x2;
typedef __attribute__((ext_vector_type(2)))  __bf16 bf16x2;

#define EMBED 64

// packed f32 -> bf16 (RNE); lowers to v_cvt_pk_bf16_f32 on gfx1250
static __device__ __forceinline__ unsigned pack2_bf16(float a, float b) {
  f32x2 fv = {a, b};
  bf16x2 hv = __builtin_convertvector(fv, bf16x2);
  return __builtin_bit_cast(unsigned, hv);
}
static __device__ __forceinline__ unsigned short f2bf1(float f) {
  __bf16 h = (__bf16)f;
  return __builtin_bit_cast(unsigned short, h);
}

// ---------------------------------------------------------------------------
// Prep: fuse M = Wo @ Wv (64x64) and c = Wo @ bv + bo for both directions.
// Weights are written to workspace ALREADY in the WMMA B-fragment layout:
//   B = M^T (32x16 tile per (nt,kc)); lane l = 16*h + n; element e <-> K = 32*kc + 16*h + e
//   Bfrag[((mat*4 + nt)*2 + kc)*32 + lane][e]  (bf16 / ushort)
// ---------------------------------------------------------------------------
__global__ __launch_bounds__(256) void ca_prep_kernel(
    const float* __restrict__ Wo1, const float* __restrict__ Wv2,
    const float* __restrict__ bv2, const float* __restrict__ bo1,
    const float* __restrict__ Wo2, const float* __restrict__ Wv1,
    const float* __restrict__ bv1, const float* __restrict__ bo2,
    unsigned short* __restrict__ Bfrag, float* __restrict__ cvec) {
  const int tid = threadIdx.x;  // 256 threads
  // fused biases: c1 (tid 0..63), c2 (tid 64..127)
  if (tid < 128) {
    const int mm = tid >> 6;
    const int j  = tid & 63;
    const float* Wo = mm ? Wo2 : Wo1;
    const float* bv = mm ? bv1 : bv2;
    const float* bo = mm ? bo2 : bo1;
    float s = bo[j];
#pragma unroll 8
    for (int i = 0; i < 64; ++i) s += Wo[j * 64 + i] * bv[i];
    cvec[mm * 64 + j] = s;
  }
  // fused weights: 2 matrices x 4096 entries, 32 per thread
  for (int t = 0; t < 32; ++t) {
    const int idx = tid * 32 + t;           // 0..8191
    const int mat = idx >> 12;              // 0: M1=Wo1@Wv2, 1: M2=Wo2@Wv1
    const int jk  = idx & 4095;
    const int j = jk >> 6, k = jk & 63;     // M[j][k]
    const float* Wo = mat ? Wo2 : Wo1;
    const float* Wv = mat ? Wv1 : Wv2;
    float s = 0.f;
#pragma unroll 8
    for (int i = 0; i < 64; ++i) s += Wo[j * 64 + i] * Wv[i * 64 + k];
    // B[k][n] = M[n][k]: n=j, place at fragment (nt,kc,lane,e)
    const int nt = j >> 4, n = j & 15;
    const int kc = k >> 5, kk = k & 31;
    const int h = kk >> 4, e = kk & 15;
    const int lane = h * 16 + n;
    Bfrag[(((mat * 4 + nt) * 2 + kc) * 32 + lane) * 16 + e] = f2bf1(s);
  }
}

// ---------------------------------------------------------------------------
// Main: 256 threads = 8 waves, 16 rows per wave. Two fused GEMMs per row tile
// via v_wmma_f32_16x16x32_bf16, then residual + layernorm in fp32.
// ---------------------------------------------------------------------------
__global__ __launch_bounds__(256) void ca_main_kernel(
    const float* __restrict__ v1, const float* __restrict__ v2,
    const unsigned short* __restrict__ Bfrag, const float* __restrict__ cvec,
    const float* __restrict__ g1, const float* __restrict__ be1,
    const float* __restrict__ g2, const float* __restrict__ be2,
    float* __restrict__ out, int nRows) {
  // A-fragment staging, pre-swizzled to WMMA layout: [wave][mat][kc][lane][e] bf16
  __shared__ __align__(32) unsigned short afrag[8][2][2][32][16];  // 32 KB

  const int tid  = threadIdx.x;
  const int wave = tid >> 5;
  const int lane = tid & 31;
  const int h    = lane >> 4;     // half-wave
  const int m    = lane & 15;     // row-in-tile (A) / col (B,C,D)
  const int rowBase = blockIdx.x * 128 + wave * 16;   // fits int (<= 262144)
  if (rowBase + 16 > nRows) return;   // wave-uniform

  // ---- stage 16 rows x 64 cols of v2 (mat0) and v1 (mat1): fp32 -> bf16,
  //      scattered into A-fragment order so consumption is one 32B ds_load.
  {
    const int prow = lane >> 1;          // producer row 0..15
    const int pcol = (lane & 1) * 32;    // col half 0 / 32
    const float* srcs[2] = { v2, v1 };   // mat0 multiplies v2; mat1 multiplies v1
#pragma unroll
    for (int mat = 0; mat < 2; ++mat) {
      const float4* src = (const float4*)(srcs[mat] + (rowBase + prow) * EMBED + pcol);
#pragma unroll
      for (int q = 0; q < 8; ++q) {
        float4 f = src[q];
        const int c  = pcol + 4 * q;
        const int kc = c >> 5, kk = c & 31;
        const int hh = kk >> 4;
        const int p0 = (kk & 15) >> 1;  // element pair idx
        uint2 pk;
        pk.x = pack2_bf16(f.x, f.y);
        pk.y = pack2_bf16(f.z, f.w);
        *(uint2*)&afrag[wave][mat][kc][hh * 16 + prow][p0 * 2] = pk;
      }
    }
  }
  // producer lanes != consumer lanes within the wave: wait for LDS stores
  asm volatile("s_wait_dscnt 0" ::: "memory");

#pragma unroll
  for (int mat = 0; mat < 2; ++mat) {
    // A fragments for this wave's 16x64 tile (one per K-chunk)
    v16bf aK0 = *(const v16bf*)&afrag[wave][mat][0][lane][0];
    v16bf aK1 = *(const v16bf*)&afrag[wave][mat][1][lane][0];

    const float* resid = (mat == 0) ? v1 : v2;   // mat0: +v1 residual
    const float* g     = (mat == 0) ? g1 : g2;
    const float* be    = (mat == 0) ? be1 : be2;
    const float* cb    = cvec + mat * 64;

    // GEMM: out_tile(16x64) = A(16x64) * B(64x64), 4 N-tiles x 2 K-chunks
    v8f acc[4];
#pragma unroll
    for (int nt = 0; nt < 4; ++nt) {
      v8f c = {};
      v16bf b0 = *(const v16bf*)&Bfrag[(((mat * 4 + nt) * 2 + 0) * 32 + lane) * 16];
      c = __builtin_amdgcn_wmma_f32_16x16x32_bf16(false, aK0, false, b0,
                                                  (short)0, c, false, false);
      v16bf b1 = *(const v16bf*)&Bfrag[(((mat * 4 + nt) * 2 + 1) * 32 + lane) * 16];
      c = __builtin_amdgcn_wmma_f32_16x16x32_bf16(false, aK1, false, b1,
                                                  (short)0, c, false, false);
      acc[nt] = c;
    }

    // per-column constants (col = nt*16 + m)
    float colc[4], colg[4], colb[4];
#pragma unroll
    for (int nt = 0; nt < 4; ++nt) {
      colc[nt] = cb[nt * 16 + m];
      colg[nt] = g[nt * 16 + m];
      colb[nt] = be[nt * 16 + m];
    }

    float* outbase = out + (size_t)mat * (size_t)nRows * EMBED;

    // D layout: VGPR r <-> row (r + 8h), lane%16 <-> col. LayerNorm per row =
    // reduce over 4 nt regs x 16 lanes of one half-wave (xor masks 1,2,4,8).
    // var = E[x^2] - mu^2 so both reductions run concurrently.
#pragma unroll
    for (int r = 0; r < 8; ++r) {
      const int row = rowBase + r + 8 * h;
      float x[4];
      float s = 0.f, s2 = 0.f;
#pragma unroll
      for (int nt = 0; nt < 4; ++nt) {
        x[nt] = acc[nt][r] + colc[nt] + resid[row * EMBED + nt * 16 + m];
        s  += x[nt];
        s2 += x[nt] * x[nt];
      }
      s  += __shfl_xor(s, 1, 32);
      s2 += __shfl_xor(s2, 1, 32);
      s  += __shfl_xor(s, 2, 32);
      s2 += __shfl_xor(s2, 2, 32);
      s  += __shfl_xor(s, 4, 32);
      s2 += __shfl_xor(s2, 4, 32);
      s  += __shfl_xor(s, 8, 32);
      s2 += __shfl_xor(s2, 8, 32);
      const float mu  = s * (1.0f / 64.0f);
      const float var = s2 * (1.0f / 64.0f) - mu * mu;
      const float rs  = rsqrtf(var + 1e-5f);
#pragma unroll
      for (int nt = 0; nt < 4; ++nt) {
        const float y = (x[nt] - mu) * rs * colg[nt] + colb[nt];
        __builtin_nontemporal_store(y, &outbase[row * EMBED + nt * 16 + m]);
      }
    }
  }
}

// ---------------------------------------------------------------------------
extern "C" void kernel_launch(void* const* d_in, const int* in_sizes, int n_in,
                              void* d_out, int out_size, void* d_ws, size_t ws_size,
                              hipStream_t stream) {
  const float* v1  = (const float*)d_in[0];
  const float* v2  = (const float*)d_in[1];
  // d_in[2..5] = Wq1,bq1,Wk2,bk2 (dead: softmax over length-1 axis == 1)
  const float* Wv2 = (const float*)d_in[6];
  const float* bv2 = (const float*)d_in[7];
  const float* Wo1 = (const float*)d_in[8];
  const float* bo1 = (const float*)d_in[9];
  // d_in[10..13] = Wq2,bq2,Wk1,bk1 (dead)
  const float* Wv1 = (const float*)d_in[14];
  const float* bv1 = (const float*)d_in[15];
  const float* Wo2 = (const float*)d_in[16];
  const float* bo2 = (const float*)d_in[17];
  const float* g1  = (const float*)d_in[18];
  const float* be1 = (const float*)d_in[19];
  const float* g2  = (const float*)d_in[20];
  const float* be2 = (const float*)d_in[21];

  const int nRows = in_sizes[0] / EMBED;

  unsigned short* Bfrag = (unsigned short*)d_ws;                       // 16 KB
  float* cvec = (float*)((char*)d_ws + 8192 * sizeof(unsigned short)); // 512 B

  ca_prep_kernel<<<1, 256, 0, stream>>>(Wo1, Wv2, bv2, bo1,
                                        Wo2, Wv1, bv1, bo2, Bfrag, cvec);

  const int blocks = (nRows + 127) / 128;  // 128 rows per block (8 waves x 16)
  ca_main_kernel<<<blocks, 256, 0, stream>>>(v1, v2, Bfrag, cvec,
                                             g1, be1, g2, be2,
                                             (float*)d_out, nRows);
}
